// HydraAttention_51625506898516
// MI455X (gfx1250) — matrix operand order repaired and checked by reading, not compiled, not run on hardware
//
#include <hip/hip_runtime.h>
#include <hip/hip_bf16.h>
#include <stdint.h>

// ---------------------------------------------------------------------------
// Hydra attention block for MI455X (gfx1250, wave32, WMMA).
//   B=4, T=4096, D=1024.  M = B*T = 16384.
//   GEMM1: qkv = x @ Wqkv + bqkv   (16384 x 1024 x 3072), bf16 WMMA, f32 acc
//   norms, kv pool, q-hat * kv
//   GEMM2: out = tmp @ Wout + bout (16384 x 1024 x 1024), bf16 WMMA, f32 acc
// GEMM uses async global->LDS copies (ASYNCcnt) with double-buffered LDS when
// the toolchain exposes the gfx1250 async builtins; otherwise falls back to
// the sync path.
// ---------------------------------------------------------------------------

typedef __bf16 bf16x16 __attribute__((ext_vector_type(16)));
typedef float  f32x8   __attribute__((ext_vector_type(8)));
typedef int    v4i     __attribute__((ext_vector_type(4)));

union Frag16 { bf16x16 v; uint32_t u[8]; };

__device__ __forceinline__ float bf2f(uint16_t h) {
    union { uint32_t u; float f; } c; c.u = ((uint32_t)h) << 16; return c.f;
}
__device__ __forceinline__ uint16_t f2bf(float f) {
    union { float f; uint32_t u; } c; c.f = f;
    uint32_t r = c.u + 0x7FFFu + ((c.u >> 16) & 1u);   // round-to-nearest-even
    return (uint16_t)(r >> 16);
}

// ------------------------- constants ---------------------------------------
#define HB   4
#define HT   4096
#define HD   1024
#define HM   (HB * HT)          // 16384 rows
#define HN1  (3 * HD)           // 3072
#define TM   128
#define TN   128
#define TK   32
#define LDT  40                 // padded LDS row stride (elements)

// ------------------------- async copy support ------------------------------
#if __has_builtin(__builtin_amdgcn_global_load_async_to_lds_b128) && \
    __has_builtin(__builtin_amdgcn_s_wait_asynccnt)
#define HYDRA_ASYNC 1
#else
#define HYDRA_ASYNC 0
#endif

#define HAS1 __attribute__((address_space(1)))
#define HAS3 __attribute__((address_space(3)))

#if HYDRA_ASYNC
__device__ __forceinline__ void hydra_async_cp16(const uint16_t* g, uint16_t* l) {
    // 16 bytes (8 bf16) per lane, global -> LDS via async engine (ASYNCcnt)
    __builtin_amdgcn_global_load_async_to_lds_b128(
        (HAS1 v4i*)g, (HAS3 v4i*)l, 0, 0);
}
__device__ __forceinline__ void hydra_wait_async() {
    __builtin_amdgcn_s_wait_asynccnt(0);
}
#endif

// ------------------------- conversion kernels ------------------------------
__global__ __launch_bounds__(256) void hydra_f32_to_bf16(
    const float* __restrict__ src, uint16_t* __restrict__ dst, int n)
{
    int g = blockIdx.x * 256 + threadIdx.x;
    if (g < n) dst[g] = f2bf(src[g]);
}

// src: R x C (row-major f32)  ->  dst: C x R (row-major bf16)  (transpose)
__global__ __launch_bounds__(256) void hydra_transpose_bf16(
    const float* __restrict__ src, uint16_t* __restrict__ dst, int R, int C)
{
    int g = blockIdx.x * 256 + threadIdx.x;
    if (g < R * C) {
        int r = g / C, c = g - r * C;              // coalesced read
        dst[(size_t)c * R + r] = f2bf(src[g]);
    }
}

// ------------------------- WMMA GEMM ---------------------------------------
// C[m,n] = sum_k A[m,k] * Bt[n,k] + bias[n]
// A : M x K bf16 row-major, Bt : N x K bf16 row-major (pre-transposed weights)
template <bool OUT_BF16>
__global__ __launch_bounds__(256) void hydra_gemm_wmma(
    const uint16_t* __restrict__ A,
    const uint16_t* __restrict__ Bt,
    const float*    __restrict__ bias,
    void*           __restrict__ Cout,
    int M, int N, int K)
{
    __shared__ __align__(16) uint16_t lA[2][TM * LDT];
    __shared__ __align__(16) uint16_t lB[2][TN * LDT];

    const int tid  = threadIdx.x;
    const int lane = tid & 31;
    const int wave = tid >> 5;        // 8 waves
    const int wm   = wave >> 1;       // 0..3  -> 32-row slab
    const int wn   = wave & 1;        // 0..1  -> 64-col slab
    const int bm   = blockIdx.y * TM;
    const int bn   = blockIdx.x * TN;

    f32x8 acc[2][4] = {};

    // global tile-load coords: each thread moves one 8-bf16 (16B) chunk,
    // two row-passes cover 128 rows x 32 cols.
    const int lc = tid & 3;           // k-chunk (8 elements each)
    const int lr = tid >> 2;          // row 0..63 (+64 on pass 2)

    const uint16_t* gA0 = A  + (size_t)(bm + lr)      * K + lc * 8;
    const uint16_t* gA1 = A  + (size_t)(bm + lr + 64) * K + lc * 8;
    const uint16_t* gB0 = Bt + (size_t)(bn + lr)      * K + lc * 8;
    const uint16_t* gB1 = Bt + (size_t)(bn + lr + 64) * K + lc * 8;
    const int sOff0 = lr * LDT + lc * 8;
    const int sOff1 = (lr + 64) * LDT + lc * 8;

    // A-fragment k map (16-bit A 16x32, ISA 7.12.2):
    //   half h = lane>>4; v0..3 -> k = 8h + 2v ; v4..7 -> k = 16 + 8h + 2(v-4)
    // B-fragment k map (16-bit B 32x16): same as A (v0..3 -> k = 8h + 2v ; v4..7 -> k = 16 + 8h + 2(v-4))
    const int h = lane >> 4;
    int kA[8], kB[8];
#pragma unroll
    for (int v = 0; v < 4; ++v) { kA[v] = 8 * h + 2 * v; kB[v] = 8 * h + 2 * v; }
#pragma unroll
    for (int v = 4; v < 8; ++v) { kA[v] = 16 + 8 * h + 2 * (v - 4); kB[v] = 16 + 8 * h + 2 * (v - 4); }
    const int frow = lane & 15;

#if HYDRA_ASYNC
    // ---------------- async double-buffered pipeline ----------------------
    // prologue: slab 0 into buffer 0
    hydra_async_cp16(gA0, &lA[0][sOff0]);
    hydra_async_cp16(gA1, &lA[0][sOff1]);
    hydra_async_cp16(gB0, &lB[0][sOff0]);
    hydra_async_cp16(gB1, &lB[0][sOff1]);

    int buf = 0;
    for (int k0 = 0; k0 < K; k0 += TK) {
        hydra_wait_async();          // my copies into lA/lB[buf] have landed
        __syncthreads();             // everyone landed; prev reads of buf^1 done

        const int nb = buf ^ 1;
        if (k0 + TK < K) {           // overlap next slab's copies with WMMA
            hydra_async_cp16(gA0 + k0 + TK, &lA[nb][sOff0]);
            hydra_async_cp16(gA1 + k0 + TK, &lA[nb][sOff1]);
            hydra_async_cp16(gB0 + k0 + TK, &lB[nb][sOff0]);
            hydra_async_cp16(gB1 + k0 + TK, &lB[nb][sOff1]);
        }

        Frag16 af[2], bfr[4];
#pragma unroll
        for (int ti = 0; ti < 2; ++ti) {
            const int r = wm * 32 + ti * 16 + frow;
#pragma unroll
            for (int v = 0; v < 8; ++v)
                af[ti].u[v] = *(const uint32_t*)&lA[buf][r * LDT + kA[v]];
        }
#pragma unroll
        for (int tj = 0; tj < 4; ++tj) {
            const int r = wn * 64 + tj * 16 + frow;
#pragma unroll
            for (int v = 0; v < 8; ++v)
                bfr[tj].u[v] = *(const uint32_t*)&lB[buf][r * LDT + kB[v]];
        }
#pragma unroll
        for (int ti = 0; ti < 2; ++ti)
#pragma unroll
            for (int tj = 0; tj < 4; ++tj)
                acc[ti][tj] = __builtin_amdgcn_wmma_f32_16x16x32_bf16(
                    false, af[ti].v, false, bfr[tj].v,
                    (short)0, acc[ti][tj], false, false);
        buf = nb;
    }
#else
    // ---------------- sync fallback -----------------------
    for (int k0 = 0; k0 < K; k0 += TK) {
        __syncthreads();
        *(uint4*)&lA[0][sOff0] = *(const uint4*)(gA0 + k0);
        *(uint4*)&lA[0][sOff1] = *(const uint4*)(gA1 + k0);
        *(uint4*)&lB[0][sOff0] = *(const uint4*)(gB0 + k0);
        *(uint4*)&lB[0][sOff1] = *(const uint4*)(gB1 + k0);
        __syncthreads();

        Frag16 af[2], bfr[4];
#pragma unroll
        for (int ti = 0; ti < 2; ++ti) {
            const int r = wm * 32 + ti * 16 + frow;
#pragma unroll
            for (int v = 0; v < 8; ++v)
                af[ti].u[v] = *(const uint32_t*)&lA[0][r * LDT + kA[v]];
        }
#pragma unroll
        for (int tj = 0; tj < 4; ++tj) {
            const int r = wn * 64 + tj * 16 + frow;
#pragma unroll
            for (int v = 0; v < 8; ++v)
                bfr[tj].u[v] = *(const uint32_t*)&lB[0][r * LDT + kB[v]];
        }
#pragma unroll
        for (int ti = 0; ti < 2; ++ti)
#pragma unroll
            for (int tj = 0; tj < 4; ++tj)
                acc[ti][tj] = __builtin_amdgcn_wmma_f32_16x16x32_bf16(
                    false, af[ti].v, false, bfr[tj].v,
                    (short)0, acc[ti][tj], false, false);
    }
#endif

    // epilogue: C/D layout (ISA 7.12.2): lane l, VGPR v -> M = v + 8*(l>>4), N = l&15
    const int m0 = bm + wm * 32 + 8 * (lane >> 4);
    const int n0 = bn + wn * 64 + (lane & 15);
#pragma unroll
    for (int tj = 0; tj < 4; ++tj) {
        const int n  = n0 + tj * 16;
        const float bv = bias[n];
#pragma unroll
        for (int ti = 0; ti < 2; ++ti) {
#pragma unroll
            for (int v = 0; v < 8; ++v) {
                const int m = m0 + ti * 16 + v;
                const float val = acc[ti][tj][v] + bv;
                if (OUT_BF16)
                    ((uint16_t*)Cout)[(size_t)m * N + n] = f2bf(val);
                else
                    ((float*)Cout)[(size_t)m * N + n] = val;
            }
        }
    }
}

// ------------------------- norms -------------------------------------------
// one block per row: qn[r] = rsqrt(sum q^2), kn[r] = rsqrt(sum k^2)
__global__ __launch_bounds__(128) void hydra_norms(
    const uint16_t* __restrict__ qkv, float* __restrict__ qn, float* __restrict__ kn)
{
    __shared__ float sq[128], sk[128];
    const int r = blockIdx.x;
    const uint16_t* row = qkv + (size_t)r * HN1;
    float aq = 0.f, ak = 0.f;
    for (int j = threadIdx.x; j < HD; j += 128) {
        float q = bf2f(row[j]);        aq += q * q;
        float k = bf2f(row[HD + j]);   ak += k * k;
    }
    sq[threadIdx.x] = aq; sk[threadIdx.x] = ak;
    __syncthreads();
    for (int s = 64; s > 0; s >>= 1) {
        if (threadIdx.x < s) {
            sq[threadIdx.x] += sq[threadIdx.x + s];
            sk[threadIdx.x] += sk[threadIdx.x + s];
        }
        __syncthreads();
    }
    if (threadIdx.x == 0) { qn[r] = rsqrtf(sq[0]); kn[r] = rsqrtf(sk[0]); }
}

// ------------------------- kv pool -----------------------------------------
__global__ __launch_bounds__(256) void hydra_zero_f32(float* __restrict__ p, int n)
{
    int g = blockIdx.x * 256 + threadIdx.x;
    if (g < n) p[g] = 0.f;
}

// kv[b,d] += sum over a T-chunk of  k_hat[b,t,d] * v[b,t,d]
__global__ __launch_bounds__(256) void hydra_kv_pool(
    const uint16_t* __restrict__ qkv, const float* __restrict__ kn,
    float* __restrict__ kv)
{
    const int g = blockIdx.x * 256 + threadIdx.x;   // 0 .. B*D-1
    const int b = g >> 10;
    const int d = g & (HD - 1);
    const int t0 = blockIdx.y * 256;                // 16 chunks of 256
    float acc = 0.f;
    for (int t = t0; t < t0 + 256; ++t) {
        const size_t row = (size_t)(b * HT + t);
        const float kh = bf2f(qkv[row * HN1 + HD + d]) * kn[row];
        const float vv = bf2f(qkv[row * HN1 + 2 * HD + d]);
        acc += kh * vv;
    }
    atomicAdd(&kv[g], acc);
}

// ------------------------- q-hat * kv --------------------------------------
__global__ __launch_bounds__(256) void hydra_scale_q(
    const uint16_t* __restrict__ qkv, const float* __restrict__ qn,
    const float* __restrict__ kv, uint16_t* __restrict__ outh)
{
    const int g = blockIdx.x * 256 + threadIdx.x;   // 0 .. M*D-1
    const int r = g >> 10;
    const int d = g & (HD - 1);
    const int b = r >> 12;                          // r / 4096
    const float q = bf2f(qkv[(size_t)r * HN1 + d]);
    outh[g] = f2bf(q * qn[r] * kv[b * HD + d]);
}

// ------------------------- host launch -------------------------------------
extern "C" void kernel_launch(void* const* d_in, const int* in_sizes, int n_in,
                              void* d_out, int out_size, void* d_ws, size_t ws_size,
                              hipStream_t stream) {
    const float* x    = (const float*)d_in[0];   // (4,4096,1024)
    const float* Wqkv = (const float*)d_in[1];   // (1024,3072)
    const float* bqkv = (const float*)d_in[2];   // (3072,)
    const float* Wout = (const float*)d_in[3];   // (1024,1024)
    const float* bout = (const float*)d_in[4];   // (1024,)
    float* out = (float*)d_out;                  // (4,4096,1024) f32

    // workspace carve-up (256B aligned)
    uint8_t* ws = (uint8_t*)d_ws;
    size_t off = 0;
    auto carve = [&](size_t bytes) { uint8_t* p = ws + off;
                                     off = (off + bytes + 255) & ~(size_t)255; return p; };
    uint16_t* xh   = (uint16_t*)carve((size_t)HM * HD * 2);    // x in bf16
    uint16_t* wt1  = (uint16_t*)carve((size_t)HN1 * HD * 2);   // Wqkv^T  (3072 x 1024)
    uint16_t* wt2  = (uint16_t*)carve((size_t)HD * HD * 2);    // Wout^T  (1024 x 1024)
    uint16_t* qkv  = (uint16_t*)carve((size_t)HM * HN1 * 2);   // (16384 x 3072) bf16
    float*    qn   = (float*)carve((size_t)HM * 4);
    float*    kn   = (float*)carve((size_t)HM * 4);
    float*    kv   = (float*)carve((size_t)HB * HD * 4);
    uint16_t* outh = (uint16_t*)carve((size_t)HM * HD * 2);    // q_hat*kv, bf16
    (void)ws_size; (void)in_sizes; (void)n_in; (void)out_size;

    // 1) precision conversion + weight transposes
    hydra_f32_to_bf16<<<(HM * HD) / 256, 256, 0, stream>>>(x, xh, HM * HD);
    hydra_transpose_bf16<<<(HD * HN1) / 256, 256, 0, stream>>>(Wqkv, wt1, HD, HN1);
    hydra_transpose_bf16<<<(HD * HD) / 256, 256, 0, stream>>>(Wout, wt2, HD, HD);

    // 2) GEMM1: qkv = x @ Wqkv + bqkv   (bf16 output)
    hydra_gemm_wmma<true><<<dim3(HN1 / TN, HM / TM), 256, 0, stream>>>(
        xh, wt1, bqkv, qkv, HM, HN1, HD);

    // 3) row norms of q and k
    hydra_norms<<<HM, 128, 0, stream>>>(qkv, qn, kn);

    // 4) kv = sum_T k_hat * v   (zero + chunked atomic accumulation)
    hydra_zero_f32<<<(HB * HD) / 256, 256, 0, stream>>>(kv, HB * HD);
    hydra_kv_pool<<<dim3((HB * HD) / 256, HT / 256), 256, 0, stream>>>(qkv, kn, kv);

    // 5) tmp = q_hat * kv (bf16)
    hydra_scale_q<<<(HM * HD) / 256, 256, 0, stream>>>(qkv, qn, kv, outh);

    // 6) GEMM2: out = tmp @ Wout + bout  (f32 output)
    hydra_gemm_wmma<false><<<dim3(HD / TN, HM / TM), 256, 0, stream>>>(
        outh, wt2, bout, out, HM, HD, HD);
}
